// HybridGCNLSTM_83580063580908
// MI455X (gfx1250) — compile-verified, matrix operand order checked by reading, not run on hardware
//
#include <hip/hip_runtime.h>
#include <hip/hip_bf16.h>
#include <math.h>

typedef __bf16 bf16;
typedef __attribute__((ext_vector_type(16))) __bf16 bf16x16;
typedef __attribute__((ext_vector_type(8)))  __bf16 bf16x8;
typedef __attribute__((ext_vector_type(8)))  float  f32x8;

#define NJ 55
#define TT 128
#define BB 64
#define HH 384
#define GSTRIDE 1536   // 4*H
#define MROWS 8192     // B*T

__device__ __forceinline__ f32x8 zero8() {
  f32x8 v;
#pragma unroll
  for (int i = 0; i < 8; ++i) v[i] = 0.0f;
  return v;
}

// Load one 16x32 bf16 WMMA fragment (A from rows of an activation matrix, or
// B from rows of a [N x K] weight matrix == columns of the logical K x N B).
// CDNA5 16-bit A layout: lane L (half = L>>4) holds k = {8*half..8*half+7} U
// {16+8*half..23+8*half}: two contiguous 16-byte runs -> two b128 loads.
__device__ __forceinline__ bf16x16 load_frag(const bf16* base, long row_stride, int lane) {
  const bf16* rowp = base + (long)(lane & 15) * row_stride;
  const int half = lane >> 4;
  bf16x8 lo = *(const bf16x8*)(rowp + 8 * half);
  bf16x8 hi = *(const bf16x8*)(rowp + 16 + 8 * half);
  return __builtin_shufflevector(lo, hi, 0,1,2,3,4,5,6,7,8,9,10,11,12,13,14,15);
}

__device__ __forceinline__ f32x8 wmma_bf16(bf16x16 a, bf16x16 b, f32x8 c) {
  return __builtin_amdgcn_wmma_f32_16x16x32_bf16(false, a, false, b, (short)0, c, false, false);
}

__device__ __forceinline__ float sigmoidf_(float x) { return 1.0f / (1.0f + expf(-x)); }
__device__ __forceinline__ float gelu_(float v) { return 0.5f * v * (1.0f + erff(v * 0.7071067811865475f)); }

// ---------------------------------------------------------------------------
// f32 -> bf16 weight conversion
// ---------------------------------------------------------------------------
__global__ void f32_to_bf16_kernel(const float* __restrict__ in, bf16* __restrict__ out, int n) {
  int i = blockIdx.x * 256 + threadIdx.x;
  if (i < n) out[i] = (bf16)in[i];
}

// ---------------------------------------------------------------------------
// Fused GCN block1 + block2 + joint-attention pooling. One workgroup = one
// (b,t) frame; all intermediates live in LDS (~86KB << 320KB WGP LDS).
// ---------------------------------------------------------------------------
__global__ __launch_bounds__(256) void gcn_frames_kernel(
    const float* __restrict__ x, const int* __restrict__ esrc, const int* __restrict__ edst, int E,
    const float* __restrict__ g1W, const float* __restrict__ g1b,
    const float* __restrict__ bn1g, const float* __restrict__ bn1b,
    const float* __restrict__ r1W, const float* __restrict__ r1b,
    const float* __restrict__ g2W, const float* __restrict__ g2b,
    const float* __restrict__ bn2g, const float* __restrict__ bn2b,
    const float* __restrict__ r2W, const float* __restrict__ r2b,
    const float* __restrict__ jaW, const float* __restrict__ jab,
    bf16* __restrict__ frames)
{
  __shared__ float sx[NJ * 4];
  __shared__ float h1[NJ * 64];
  __shared__ float a1[NJ * 64];     // agg1, then reused as y1
  __shared__ float h2[NJ * 128];    // lin2, then reused as y2
  __shared__ float a2[NJ * 128];
  __shared__ float deg[NJ];
  __shared__ float nrm[128];
  __shared__ float lg[NJ];

  const int f = blockIdx.x;
  const int tid = threadIdx.x;
  const float* xf = x + (size_t)f * NJ * 4;

  for (int i = tid; i < NJ * 4; i += 256) sx[i] = xf[i];
  for (int i = tid; i < NJ; i += 256) deg[i] = 1.0f;   // self loop
  for (int i = tid; i < NJ * 64; i += 256) a1[i] = 0.0f;
  for (int i = tid; i < NJ * 128; i += 256) a2[i] = 0.0f;
  __syncthreads();
  for (int e = tid; e < E; e += 256) atomicAdd(&deg[edst[e]], 1.0f);
  __syncthreads();
  for (int e = tid; e < E; e += 256) nrm[e] = rsqrtf(deg[esrc[e]]) * rsqrtf(deg[edst[e]]);

  // ---- block 1 (4 -> 64) ----
  for (int i = tid; i < NJ * 64; i += 256) {
    int n = i >> 6, c = i & 63;
    float s = 0.0f;
#pragma unroll
    for (int k = 0; k < 4; ++k) s += sx[n * 4 + k] * g1W[k * 64 + c];
    h1[i] = s;
  }
  __syncthreads();
  for (int i = tid; i < E * 64; i += 256) {
    int e = i >> 6, c = i & 63;
    atomicAdd(&a1[edst[e] * 64 + c], h1[esrc[e] * 64 + c] * nrm[e]);
  }
  __syncthreads();
  const float bnc = rsqrtf(1.0f + 1e-5f);
  for (int i = tid; i < NJ * 64; i += 256) {
    int n = i >> 6, c = i & 63;
    float v = (a1[i] + h1[i] / deg[n] + g1b[c]) * (bn1g[c] * bnc) + bn1b[c];
    float r = r1b[c];
#pragma unroll
    for (int k = 0; k < 4; ++k) r += sx[n * 4 + k] * r1W[k * 64 + c];
    a1[i] = gelu_(v + r);                 // a1 := y1
  }
  __syncthreads();

  // ---- block 2 (64 -> 128) ----
  for (int i = tid; i < NJ * 128; i += 256) {
    int n = i >> 7, c = i & 127;
    float s = 0.0f;
    for (int k = 0; k < 64; ++k) s += a1[n * 64 + k] * g2W[k * 128 + c];
    h2[i] = s;
  }
  __syncthreads();
  for (int i = tid; i < E * 128; i += 256) {
    int e = i >> 7, c = i & 127;
    atomicAdd(&a2[edst[e] * 128 + c], h2[esrc[e] * 128 + c] * nrm[e]);
  }
  __syncthreads();
  for (int i = tid; i < NJ * 128; i += 256) {
    int n = i >> 7, c = i & 127;
    float v = (a2[i] + h2[i] / deg[n] + g2b[c]) * (bn2g[c] * bnc) + bn2b[c];
    float r = r2b[c];
    for (int k = 0; k < 64; ++k) r += a1[n * 64 + k] * r2W[k * 128 + c];
    h2[i] = gelu_(v + r);                 // h2 := y2
  }
  __syncthreads();

  // ---- joint attention pooling over N ----
  if (tid < NJ) {
    float s = jab[0];
    for (int c = 0; c < 128; ++c) s += h2[tid * 128 + c] * jaW[c];
    lg[tid] = s;
  }
  __syncthreads();
  if (tid == 0) {
    float m = -1e30f;
    for (int n = 0; n < NJ; ++n) m = fmaxf(m, lg[n]);
    float su = 0.0f;
    for (int n = 0; n < NJ; ++n) { lg[n] = expf(lg[n] - m); su += lg[n]; }
    float inv = 1.0f / su;
    for (int n = 0; n < NJ; ++n) lg[n] *= inv;
  }
  __syncthreads();
  if (tid < 128) {
    float s = 0.0f;
    for (int n = 0; n < NJ; ++n) s += h2[n * 128 + tid] * lg[n];
    frames[(size_t)f * 128 + tid] = (bf16)s;
  }
}

// ---------------------------------------------------------------------------
// Batch-parallel input projection GEMM with WMMA:
//   xg[dir][t][n][b] = sum_k x[b*T+t][k] * Wih[dir][n][k] + bih[dir][n] + bhh[dir][n]
// M=8192 rows x N=1536 x K=IN, 2 directions. One wave = 16x64 output tile
// (A fragment reused across 4 WMMAs). Output is stored BATCH-INNERMOST so the
// serial LSTM scan can init each gate accumulator with a single b128 load.
// ---------------------------------------------------------------------------
__global__ __launch_bounds__(256) void xg_gemm_kernel(
    const bf16* __restrict__ x, int IN,
    const bf16* __restrict__ Wih,                                   // [2][1536][IN]
    const float* __restrict__ bih, const float* __restrict__ bhh,   // [2][1536]
    bf16* __restrict__ xg)                                          // [2][T][1536][64]
{
  const int tid = threadIdx.x;
  const int wave = tid >> 5, lane = tid & 31;
  const int col = lane & 15;
  const int rhalf = (lane >> 4) << 3;

  const int job = blockIdx.x * 8 + wave;       // 2 * 512 * 24 = 24576 jobs
  const int dir = job / (512 * 24);
  const int rem = job % (512 * 24);
  const int m0 = (rem / 24) * 16;
  const int n0 = (rem % 24) * 64;

  const bf16* WihD = Wih + (size_t)dir * GSTRIDE * IN;

  f32x8 acc[4];
#pragma unroll
  for (int j = 0; j < 4; ++j) acc[j] = zero8();

  for (int k0 = 0; k0 < IN; k0 += 32) {
    bf16x16 a = load_frag(x + (size_t)m0 * IN + k0, IN, lane);
#pragma unroll
    for (int j = 0; j < 4; ++j) {
      bf16x16 b = load_frag(WihD + (size_t)(n0 + j * 16) * IN + k0, IN, lane);
      acc[j] = wmma_bf16(a, b, acc[j]);
    }
  }

  const float* bihD = bih + dir * GSTRIDE;
  const float* bhhD = bhh + dir * GSTRIDE;
  bf16* xgD = xg + (size_t)dir * TT * GSTRIDE * BB;
  const int bidx = m0 >> 7;                     // batch index (constant per tile)
#pragma unroll
  for (int j = 0; j < 4; ++j) {
    const int n = n0 + j * 16 + col;
    const float bsum = bihD[n] + bhhD[n];
#pragma unroll
    for (int r = 0; r < 8; ++r) {
      const int t = (m0 & (TT - 1)) + rhalf + r;  // time index within batch row
      xgD[((size_t)t * GSTRIDE + n) * BB + bidx] = (bf16)(acc[j][r] + bsum);
    }
  }
}

// ---------------------------------------------------------------------------
// Recurrent-only BiLSTM scan. One workgroup per direction, 512 threads =
// 16 waves. Per-step 64x1536 gate GEMM (K=384 only) = 96 tile-pairs x 4
// gates; wave w processes pairs {6w..6w+5} SEQUENTIALLY so only one pair's
// 4 gate accumulators (32 VGPRs) are live at a time. Accumulators initialize
// straight from batch-innermost xg with one b128 load per gate. The i/f/g/o
// accumulators of a cell sit in the same lane, so c-state (48 VGPRs) never
// leaves registers. h is double-buffered in LDS as bf16 (2 x 48KB), one
// barrier per step.
// ---------------------------------------------------------------------------
__global__ __launch_bounds__(512) void lstm_scan_kernel(
    const bf16* __restrict__ xg,    // [2][T][1536][64] input proj + biases
    const bf16* __restrict__ Whh,   // [2][1536][384]
    bf16* __restrict__ out_bf, float* __restrict__ out_f32)
{
  __shared__ __align__(16) bf16 hsbuf[2][BB * HH];   // 2 x 48KB
  const int dir = blockIdx.x;                        // 0 = fwd, 1 = bwd
  const bf16* xgD  = xg + (size_t)dir * TT * GSTRIDE * BB;
  const bf16* WhhD = Whh + (size_t)dir * GSTRIDE * HH;

  const int tid = threadIdx.x;
  const int wave = tid >> 5, lane = tid & 31;
  const int col = lane & 15;
  const int rhalf = (lane >> 4) << 3;

  for (int i = tid; i < BB * HH; i += 512) hsbuf[0][i] = (bf16)0.0f;

  f32x8 cst[6];
#pragma unroll
  for (int q = 0; q < 6; ++q) cst[q] = zero8();

  __syncthreads();

  for (int t = 0; t < TT; ++t) {
    const int tt = dir ? (TT - 1 - t) : t;
    const int cur = t & 1, nxt = cur ^ 1;
    const bf16* xgT = xgD + (size_t)tt * GSTRIDE * BB;

#pragma unroll
    for (int q = 0; q < 6; ++q) {
      const int p = wave * 6 + q;
      const int m0 = (p / 24) * 16;
      const int n0 = (p % 24) * 16;
      const int u = n0 + col;

      // init accumulators from precomputed input projection (biases folded):
      // per gate, lane's 8 C elements are 8 consecutive batch idxs -> 1x b128
      f32x8 acc[4];
#pragma unroll
      for (int g = 0; g < 4; ++g) {
        const bf16x8 v = *(const bf16x8*)(xgT + (size_t)(g * HH + u) * BB + m0 + rhalf);
#pragma unroll
        for (int r = 0; r < 8; ++r) acc[g][r] = (float)v[r];
      }

      // recurrent projection: gates += h @ Whh^T (h in LDS, bf16)
      for (int k0 = 0; k0 < HH; k0 += 32) {
        bf16x16 a = load_frag(&hsbuf[cur][m0 * HH + k0], HH, lane);
#pragma unroll
        for (int g = 0; g < 4; ++g) {
          const bf16* wrow = WhhD + (size_t)(g * HH + n0) * HH + k0;
          __builtin_prefetch((const void*)(wrow + 64), 0, 0);  // warm next k-step line
          bf16x16 b = load_frag(wrow, HH, lane);
          acc[g] = wmma_bf16(a, b, acc[g]);
        }
      }

      // LSTM cell update; write new h into the OTHER LDS buffer
#pragma unroll
      for (int r = 0; r < 8; ++r) {
        const int bidx = m0 + rhalf + r;             // C/D: M = r + 8*half
        const float gi = sigmoidf_(acc[0][r]);
        const float gf = sigmoidf_(acc[1][r]);
        const float gg = tanhf(acc[2][r]);
        const float go = sigmoidf_(acc[3][r]);
        const float c  = gf * cst[q][r] + gi * gg;
        cst[q][r] = c;
        const float h = go * tanhf(c);
        hsbuf[nxt][bidx * HH + u] = (bf16)h;
        const size_t o = ((size_t)bidx * TT + tt) * 768 + (size_t)dir * HH + u;
        if (out_bf)  out_bf[o]  = (bf16)h;
        if (out_f32) out_f32[o] = h;
      }
    }
    __syncthreads();   // new h complete before next step reads it
  }
}

// ---------------------------------------------------------------------------
// Temporal attention pooling over T (one block per batch row)
// ---------------------------------------------------------------------------
__global__ __launch_bounds__(128) void temporal_pool_kernel(
    const float* __restrict__ seq, const float* __restrict__ taW,
    const float* __restrict__ tab, float* __restrict__ pooled)
{
  __shared__ float lg[TT];
  const int b = blockIdx.x, tid = threadIdx.x;
  const float* s = seq + (size_t)b * TT * 768;
  float dot = tab[0];
  for (int i = 0; i < 768; ++i) dot += s[(size_t)tid * 768 + i] * taW[i];
  lg[tid] = dot;
  __syncthreads();
  if (tid == 0) {
    float m = -1e30f;
    for (int t = 0; t < TT; ++t) m = fmaxf(m, lg[t]);
    float su = 0.0f;
    for (int t = 0; t < TT; ++t) { lg[t] = expf(lg[t] - m); su += lg[t]; }
    float inv = 1.0f / su;
    for (int t = 0; t < TT; ++t) lg[t] *= inv;
  }
  __syncthreads();
  for (int c = tid; c < 768; c += 128) {
    float acc = 0.0f;
    for (int t = 0; t < TT; ++t) acc += s[(size_t)t * 768 + c] * lg[t];
    pooled[(size_t)b * 768 + c] = acc;
  }
}

// ---------------------------------------------------------------------------
// LayerNorm -> fc1 (768->256) GELU -> fc2 (256->100). One block per row.
// ---------------------------------------------------------------------------
__global__ __launch_bounds__(256) void classifier_kernel(
    const float* __restrict__ pooled, const float* __restrict__ lng, const float* __restrict__ lnb,
    const float* __restrict__ fc1W, const float* __restrict__ fc1b,
    const float* __restrict__ fc2W, const float* __restrict__ fc2b,
    float* __restrict__ out)
{
  __shared__ float z[768];
  __shared__ float hh[256];
  __shared__ float mu_s, rstd_s;
  const int b = blockIdx.x, tid = threadIdx.x;
  const float* p = pooled + (size_t)b * 768;
  for (int i = tid; i < 768; i += 256) z[i] = p[i];
  __syncthreads();
  if (tid == 0) {
    float m = 0.0f;
    for (int i = 0; i < 768; ++i) m += z[i];
    m *= (1.0f / 768.0f);
    float v = 0.0f;
    for (int i = 0; i < 768; ++i) { float d = z[i] - m; v += d * d; }
    v *= (1.0f / 768.0f);
    mu_s = m; rstd_s = rsqrtf(v + 1e-5f);
  }
  __syncthreads();
  for (int i = tid; i < 768; i += 256) z[i] = (z[i] - mu_s) * rstd_s * lng[i] + lnb[i];
  __syncthreads();
  {
    float a = fc1b[tid];
    for (int i = 0; i < 768; ++i) a += z[i] * fc1W[i * 256 + tid];
    hh[tid] = gelu_(a);
  }
  __syncthreads();
  if (tid < 100) {
    float o = fc2b[tid];
    for (int i = 0; i < 256; ++i) o += hh[i] * fc2W[i * 100 + tid];
    out[(size_t)b * 100 + tid] = o;
  }
}

// ---------------------------------------------------------------------------
extern "C" void kernel_launch(void* const* d_in, const int* in_sizes, int n_in,
                              void* d_out, int out_size, void* d_ws, size_t ws_size,
                              hipStream_t stream) {
  const float* x    = (const float*)d_in[0];
  const int*   esrc = (const int*)d_in[1];
  const int*   edst = (const int*)d_in[2];
  const float* g1W  = (const float*)d_in[3];
  const float* g1b  = (const float*)d_in[4];
  const float* bn1g = (const float*)d_in[5];
  const float* bn1b = (const float*)d_in[6];
  const float* r1W  = (const float*)d_in[7];
  const float* r1b  = (const float*)d_in[8];
  const float* g2W  = (const float*)d_in[9];
  const float* g2b  = (const float*)d_in[10];
  const float* bn2g = (const float*)d_in[11];
  const float* bn2b = (const float*)d_in[12];
  const float* r2W  = (const float*)d_in[13];
  const float* r2b  = (const float*)d_in[14];
  const float* jaW  = (const float*)d_in[15];
  const float* jab  = (const float*)d_in[16];
  const float* l0_Wih = (const float*)d_in[17];
  const float* l0_Whh = (const float*)d_in[18];
  const float* l0_bih = (const float*)d_in[19];
  const float* l0_bhh = (const float*)d_in[20];
  const float* l1_Wih = (const float*)d_in[21];
  const float* l1_Whh = (const float*)d_in[22];
  const float* l1_bih = (const float*)d_in[23];
  const float* l1_bhh = (const float*)d_in[24];
  const float* taW  = (const float*)d_in[25];
  const float* tab  = (const float*)d_in[26];
  const float* lng  = (const float*)d_in[27];
  const float* lnb  = (const float*)d_in[28];
  const float* fc1W = (const float*)d_in[29];
  const float* fc1b = (const float*)d_in[30];
  const float* fc2W = (const float*)d_in[31];
  const float* fc2b = (const float*)d_in[32];

  char* ws = (char*)d_ws;
  bf16*  frames = (bf16*)(ws + 0);            //  2,097,152 B
  bf16*  seq0   = (bf16*)(ws + 2097152);      // 12,582,912 B
  float* seq1   = (float*)(ws + 14680064);    // 25,165,824 B
  float* pooled = (float*)(ws + 39845888);    //    196,608 B
  bf16*  wih0   = (bf16*)(ws + 40042496);     //    786,432 B
  bf16*  whh0   = (bf16*)(ws + 40828928);     //  2,359,296 B
  bf16*  wih1   = (bf16*)(ws + 43188224);     //  4,718,592 B
  bf16*  whh1   = (bf16*)(ws + 47906816);     //  2,359,296 B
  bf16*  xgbuf  = (bf16*)(ws + 50266112);     // 50,331,648 B (reused by both layers)

  const int E = in_sizes[1];

  auto cvt = [&](const float* s, bf16* d, int n) {
    f32_to_bf16_kernel<<<(n + 255) / 256, 256, 0, stream>>>(s, d, n);
  };
  cvt(l0_Wih, wih0, 2 * GSTRIDE * 128);
  cvt(l0_Whh, whh0, 2 * GSTRIDE * HH);
  cvt(l1_Wih, wih1, 2 * GSTRIDE * 768);
  cvt(l1_Whh, whh1, 2 * GSTRIDE * HH);

  gcn_frames_kernel<<<BB * TT, 256, 0, stream>>>(
      x, esrc, edst, E, g1W, g1b, bn1g, bn1b, r1W, r1b,
      g2W, g2b, bn2g, bn2b, r2W, r2b, jaW, jab, frames);

  // ---- BiLSTM layer 0 ----
  xg_gemm_kernel<<<3072, 256, 0, stream>>>(frames, 128, wih0, l0_bih, l0_bhh, xgbuf);
  lstm_scan_kernel<<<2, 512, 0, stream>>>(xgbuf, whh0, seq0, (float*)nullptr);

  // ---- BiLSTM layer 1 ----
  xg_gemm_kernel<<<3072, 256, 0, stream>>>(seq0, 768, wih1, l1_bih, l1_bhh, xgbuf);
  lstm_scan_kernel<<<2, 512, 0, stream>>>(xgbuf, whh1, (bf16*)nullptr, seq1);

  temporal_pool_kernel<<<BB, 128, 0, stream>>>(seq1, taW, tab, pooled);
  classifier_kernel<<<BB, 256, 0, stream>>>(pooled, lng, lnb, fc1W, fc1b,
                                            fc2W, fc2b, (float*)d_out);
}
